// OpticalFlow_39822936768679
// MI455X (gfx1250) — matrix operand order repaired and checked by reading, not compiled, not run on hardware
//
#include <hip/hip_runtime.h>

typedef __attribute__((ext_vector_type(16))) _Float16 v16h;
typedef __attribute__((ext_vector_type(8)))  float    v8f;
typedef __attribute__((ext_vector_type(4)))  unsigned int u32x4;
typedef __attribute__((ext_vector_type(8)))  int      i32x8;
typedef __attribute__((ext_vector_type(4)))  int      i32x4;

#define HD_ 96
#define CD_ 64
#define ITERS_ 20

// ---- Tensor Data Mover availability (per-pass: host pass lacks the builtin
// and falls back to manual LDS staging; device pass uses TDM).
#if __has_builtin(__builtin_amdgcn_tensor_load_to_lds)
#define HAVE_TDM 1
#else
#define HAVE_TDM 0
#endif
#if __has_include(<hip/amd_detail/amd_gfx1250_TDM.h>)
#define TDM_SIX_ARGS 1   // amdgpu-toolchain (clang-23, therock headers)
#else
#define TDM_SIX_ARGS 0   // ROCm 7.2 (clang-22): 5-arg builtin
#endif

#if HAVE_TDM
// 2-D f16 tile DMA: tile_d1 rows of tile_d0 contiguous elements, row stride
// row_stride (elements), packed row-major into LDS at lds_off.
// D# packing per CDNA5 ISA 8.3/8.4: group0 = {flags, lds_addr, global_addr,
// type=2}; group1 = {data_size=2B, tensor dims/strides, tile dims}.
__device__ __forceinline__ void tdm_load_2d_f16(const _Float16* gsrc,
                                                unsigned lds_off,
                                                unsigned tensor_d0,
                                                unsigned tensor_d1,
                                                unsigned tile_d0,
                                                unsigned tile_d1,
                                                unsigned long long row_stride)
{
  unsigned long long ga = (unsigned long long)(size_t)gsrc;
  u32x4 g0 = {0u, 0u, 0u, 0u};
  g0[0] = 1u;                                            // count=1, user mode
  g0[1] = lds_off;                                       // lds_addr (bytes)
  g0[2] = (unsigned)(ga & 0xffffffffu);                  // global_addr[31:0]
  g0[3] = (unsigned)((ga >> 32) & 0x01ffffffu)           // global_addr[56:32]
        | 0x80000000u;                                   // type=2 ("image")
  i32x8 g1 = {0, 0, 0, 0, 0, 0, 0, 0};
  g1[0] = (int)(1u << 16);                               // data_size=1 -> 2B
  g1[1] = (int)((tensor_d0 & 0xffffu) << 16);            // tensor_dim0[15:0]
  g1[2] = (int)((tensor_d0 >> 16) | ((tensor_d1 & 0xffffu) << 16));
  g1[3] = (int)((tensor_d1 >> 16) | (tile_d0 << 16));    // tile_dim0
  g1[4] = (int)(tile_d1 & 0xffffu);                      // tile_dim1 (tile_dim2=0)
  g1[5] = (int)(unsigned)(row_stride & 0xffffffffu);     // dim0_stride[31:0]
  g1[6] = (int)(unsigned)((row_stride >> 32) & 0xffffu); // dim0_stride[47:32]
  i32x4 z4 = {0, 0, 0, 0};
#if TDM_SIX_ARGS
  i32x8 z8 = {0, 0, 0, 0, 0, 0, 0, 0};
  __builtin_amdgcn_tensor_load_to_lds(g0, g1, z4, z4, z8, 0);
#else
  __builtin_amdgcn_tensor_load_to_lds(g0, g1, z4, z4, 0);
#endif
}
#endif  // HAVE_TDM

// ---------------------------------------------------------------------------
// Fragment packing for V_WMMA_F32_16X16X32_F16 (wave32), per ISA 7.12.2:
//  A (16x32 f16): lane L holds row M=L&15. For VGPR v (halves h=0/1):
//    K = (v>=4?16:0) + (v&3)*2 + (L>=16?8:0) + h
//  B (32x16 f16): lane L holds col N=L&15. For VGPR v:
//    K = (v>=4?16:0) + (v&3)*2 + (L>=16?8:0) + h
//  C/D (16x16 f32): VGPR r holds (M = r + (L>=16?8:0), N = L&15)
// ---------------------------------------------------------------------------
// A stored m-major As[32][64] (conv kernel)
#define WMMA_STEP(As, Bs, am, bn, hi, kc, acc)                                \
  {                                                                           \
    v16h a_, b_;                                                              \
    _Pragma("unroll")                                                         \
    for (int v_ = 0; v_ < 8; ++v_) {                                          \
      int ka_ = (kc) + ((v_ >= 4) ? 16 : 0) + ((v_ & 3) << 1) + ((hi) ? 8 : 0); \
      a_[2 * v_]     = As[(am)][ka_];                                         \
      a_[2 * v_ + 1] = As[(am)][ka_ + 1];                                     \
      int kb_ = (kc) + ((v_ >= 4) ? 16 : 0) + ((v_ & 3) << 1) + ((hi) ? 8 : 0); \
      b_[2 * v_]     = Bs[kb_][(bn)];                                         \
      b_[2 * v_ + 1] = Bs[kb_ + 1][(bn)];                                     \
    }                                                                         \
    acc = __builtin_amdgcn_wmma_f32_16x16x32_f16(                             \
        false, a_, false, b_, (short)0, acc, false, false);                   \
  }

// A stored k-major As[64][32] (corr kernel; matches TDM row-major tile fill)
#define WMMA_STEP_T(As, Bs, am, bn, hi, kc, acc)                              \
  {                                                                           \
    v16h a_, b_;                                                              \
    _Pragma("unroll")                                                         \
    for (int v_ = 0; v_ < 8; ++v_) {                                          \
      int ka_ = (kc) + ((v_ >= 4) ? 16 : 0) + ((v_ & 3) << 1) + ((hi) ? 8 : 0); \
      a_[2 * v_]     = As[ka_][(am)];                                         \
      a_[2 * v_ + 1] = As[ka_ + 1][(am)];                                     \
      int kb_ = (kc) + ((v_ >= 4) ? 16 : 0) + ((v_ & 3) << 1) + ((hi) ? 8 : 0); \
      b_[2 * v_]     = Bs[kb_][(bn)];                                         \
      b_[2 * v_ + 1] = Bs[kb_ + 1][(bn)];                                     \
    }                                                                         \
    acc = __builtin_amdgcn_wmma_f32_16x16x32_f16(                             \
        false, a_, false, b_, (short)0, acc, false, false);                   \
  }

// ---------------------------------------------------------------------------
// Generic implicit-GEMM convolution, fused bias + activation.
//   out[Cout][Ho*Wo] = act( W[Cout][Cin*KH*KW] x im2col + bias )
// Input channels c: c < C0 -> in0[c], else in1[c-C0] (free concat).
// Block = 256 threads (8 waves); block tile = 32 Cout x 64 pixels;
// wave (mh, nq) owns one 16x16 WMMA tile; K-chunk = 64 with both 32-wide
// sub-chunks unrolled.  act: 0=none 1=relu 2=sigmoid 3=tanh
// ---------------------------------------------------------------------------
__global__ __launch_bounds__(256)
void k_conv_wmma(const float* __restrict__ in0, int C0,
                 const float* __restrict__ in1,
                 const float* __restrict__ w, const float* __restrict__ bias,
                 float* __restrict__ out,
                 int Cin, int Cout, int KH, int KW,
                 int H, int W, int Ho, int Wo,
                 int stride, int padT, int padL, int act)
{
  const int HWo  = Ho * Wo;
  const int Ktot = Cin * KH * KW;
  const int Kc   = (Ktot + 63) & ~63;
  const int pixBase = blockIdx.x * 64;
  const int coBase  = blockIdx.y * 32;
  const int tid  = threadIdx.x;
  const int lane = tid & 31;
  const int wv   = tid >> 5;
  const int mh   = wv >> 2;
  const int nq   = wv & 3;
  const int hi   = (lane >= 16) ? 1 : 0;
  const int l15  = lane & 15;
  const int am   = mh * 16 + l15;
  const int bn   = nq * 16 + l15;
  const int KK   = KH * KW;

  __shared__ _Float16 As[32][64];
  __shared__ _Float16 Bs[64][64];

  v8f acc = {};
  for (int k0 = 0; k0 < Kc; k0 += 64) {
    for (int i = tid; i < 32 * 64; i += 256) {
      int m = i >> 6, kk = i & 63;
      int k = k0 + kk;
      int co = coBase + m;
      float v = 0.f;
      if (k < Ktot && co < Cout) v = w[(size_t)co * Ktot + k];
      As[m][kk] = (_Float16)v;
    }
    for (int i = tid; i < 64 * 64; i += 256) {
      int kk = i >> 6, p = i & 63;
      int k = k0 + kk;
      float v = 0.f;
      if (k < Ktot) {
        int c  = k / KK;
        int r  = k - c * KK;
        int ky = r / KW;
        int kx = r - ky * KW;
        int pix = pixBase + p;
        int oy = pix / Wo;
        int ox = pix - oy * Wo;
        int iy = oy * stride + ky - padT;
        int ix = ox * stride + kx - padL;
        if (iy >= 0 && iy < H && ix >= 0 && ix < W) {
          const float* src = (c < C0) ? (in0 + (size_t)c * H * W)
                                      : (in1 + (size_t)(c - C0) * H * W);
          v = src[iy * W + ix];
        }
      }
      Bs[kk][p] = (_Float16)v;
    }
    __syncthreads();
    WMMA_STEP(As, Bs, am, bn, hi, 0,  acc);
    WMMA_STEP(As, Bs, am, bn, hi, 32, acc);
    __syncthreads();
  }

  const int pix = pixBase + bn;
  #pragma unroll
  for (int r = 0; r < 8; ++r) {
    int co = coBase + mh * 16 + r + hi * 8;
    if (co < Cout) {
      float v = acc[r] + bias[co];
      if (act == 1)      v = fmaxf(v, 0.f);
      else if (act == 2) v = 1.f / (1.f + __expf(-v));
      else if (act == 3) v = tanhf(v);
      out[(size_t)co * HWo + pix] = v;
    }
  }
}

// ---------------------------------------------------------------------------
// Correlation volume GEMM: corr[m][n] = (f16) scale * sum_c f1[c][m]*f2[c][n]
// K = 128 (compile-time) -> fully unrolled, 4 WMMA per wave.
// f16 output so the whole 4-level pyramid (~100 MB) stays L2-resident.
// A/B tiles are strided 2-D tiles of f16 [128][HW] -> staged by the Tensor
// Data Mover (one TENSOR_LOAD_TO_LDS per tile, TENSORcnt-synchronized);
// falls back to manual LDS staging where the builtin is unavailable.
// ---------------------------------------------------------------------------
#define CORR_C 128

__global__ __launch_bounds__(256)
void k_corr_wmma(const _Float16* __restrict__ f1,
                 const _Float16* __restrict__ f2,
                 _Float16* __restrict__ corr, int HW, float scale)
{
  const int nBase = blockIdx.x * 64;
  const int mBase = blockIdx.y * 32;
  const int tid  = threadIdx.x;
  const int lane = tid & 31;
  const int wv   = tid >> 5;
  const int mh   = wv >> 2;
  const int nq   = wv & 3;
  const int hi   = (lane >= 16) ? 1 : 0;
  const int l15  = lane & 15;
  const int am   = mh * 16 + l15;
  const int bn   = nq * 16 + l15;

  __shared__ _Float16 As[64][32];   // k-major (TDM row-major tile fill)
  __shared__ _Float16 Bs[64][64];   // k-major

  v8f acc = {};
  #pragma unroll
  for (int k0 = 0; k0 < CORR_C; k0 += 64) {
#if HAVE_TDM
    if (tid == 0) {
      unsigned ldsA = (unsigned)(size_t)&As[0][0];   // generic ptr low 32 bits
      unsigned ldsB = (unsigned)(size_t)&Bs[0][0];   //   = LDS byte address
      tdm_load_2d_f16(f1 + (size_t)k0 * HW + mBase, ldsA,
                      (unsigned)HW, CORR_C, 32, 64, (unsigned long long)HW);
      tdm_load_2d_f16(f2 + (size_t)k0 * HW + nBase, ldsB,
                      (unsigned)HW, CORR_C, 64, 64, (unsigned long long)HW);
      __builtin_amdgcn_s_wait_tensorcnt(0);
    }
#else
    if (k0 + 64 < CORR_C) {
      __builtin_prefetch(f1 + (size_t)(k0 + 64) * HW + mBase, 0, 1);
      __builtin_prefetch(f2 + (size_t)(k0 + 64) * HW + nBase, 0, 1);
    }
    for (int i = tid; i < 64 * 32; i += 256) {
      int kk = i >> 5, m = i & 31;
      As[kk][m] = f1[(size_t)(k0 + kk) * HW + mBase + m];
    }
    for (int i = tid; i < 64 * 64; i += 256) {
      int kk = i >> 6, p = i & 63;
      Bs[kk][p] = f2[(size_t)(k0 + kk) * HW + nBase + p];
    }
#endif
    __syncthreads();
    WMMA_STEP_T(As, Bs, am, bn, hi, 0,  acc);
    WMMA_STEP_T(As, Bs, am, bn, hi, 32, acc);
    __syncthreads();
  }
  const int n = nBase + bn;
  #pragma unroll
  for (int r = 0; r < 8; ++r) {
    int m = mBase + mh * 16 + r + hi * 8;
    corr[(size_t)m * HW + n] = (_Float16)(acc[r] * scale);
  }
}

// ---------------------------------------------------------------------------
// Elementwise / glue kernels
// ---------------------------------------------------------------------------
__global__ void k_preproc(const float* __restrict__ in, float* __restrict__ out, int n) {
  int i = blockIdx.x * 256 + threadIdx.x;
  if (i < n) {
    float v = in[i];
    v = fminf(fmaxf(v, 0.f), 1.f);
    out[i] = 2.f * v - 1.f;
  }
}

__global__ void k_f32_to_f16(const float* __restrict__ in, _Float16* __restrict__ out, int n) {
  int i = blockIdx.x * 256 + threadIdx.x;
  if (i < n) out[i] = (_Float16)in[i];
}

__global__ void k_net_ctx(const float* __restrict__ cmap, float* __restrict__ net,
                          float* __restrict__ ctx, int HW) {
  int i = blockIdx.x * 256 + threadIdx.x;
  int total = (HD_ + CD_) * HW;
  if (i >= total) return;
  int c = i / HW, p = i - c * HW;
  float v = cmap[i];
  if (c < HD_) net[c * HW + p] = tanhf(v);
  else         ctx[(c - HD_) * HW + p] = fmaxf(v, 0.f);
}

__global__ void k_initcoords(float* __restrict__ coords, int Wo, int HW) {
  int p = blockIdx.x * 256 + threadIdx.x;
  if (p < HW) {
    coords[p]      = (float)(p % Wo);
    coords[HW + p] = (float)(p / Wo);
  }
}

__global__ void k_flow(const float* __restrict__ coords, float* __restrict__ flow,
                       int Wo, int HW) {
  int p = blockIdx.x * 256 + threadIdx.x;
  if (p < HW) {
    flow[p]      = coords[p]      - (float)(p % Wo);
    flow[HW + p] = coords[HW + p] - (float)(p / Wo);
  }
}

__global__ void k_rmul(const float* __restrict__ r, const float* __restrict__ net,
                       float* __restrict__ dst, int n) {
  int i = blockIdx.x * 256 + threadIdx.x;
  if (i < n) dst[i] = r[i] * net[i];
}

__global__ void k_gru(float* __restrict__ net, const float* __restrict__ z,
                      const float* __restrict__ q, int n) {
  int i = blockIdx.x * 256 + threadIdx.x;
  if (i < n) {
    float zi = z[i];
    net[i] = (1.f - zi) * net[i] + zi * q[i];
  }
}

__global__ void k_addinto(float* __restrict__ dst, const float* __restrict__ d, int n) {
  int i = blockIdx.x * 256 + threadIdx.x;
  if (i < n) dst[i] += d[i];
}

__global__ void k_avgpool2h(const _Float16* __restrict__ in, _Float16* __restrict__ out,
                            int M, int h, int w) {
  int h2 = h >> 1, w2 = w >> 1;
  int n = M * h2 * w2;
  int i = blockIdx.x * 256 + threadIdx.x;
  if (i >= n) return;
  int p = i / (h2 * w2);
  int r = i - p * (h2 * w2);
  int y = r / w2, x = r - y * w2;
  const _Float16* b = in + ((size_t)p * h + 2 * y) * w + 2 * x;
  float s = (float)b[0] + (float)b[1] + (float)b[w] + (float)b[w + 1];
  out[i] = (_Float16)(0.25f * s);
}

__device__ __forceinline__ float samp_zero_h(const _Float16* img, float xi, float yi,
                                             int hl, int wl) {
  if (xi < 0.f || xi > (float)(wl - 1) || yi < 0.f || yi > (float)(hl - 1)) return 0.f;
  return (float)img[(int)yi * wl + (int)xi];
}

// feat[j][p], j = l*49 + (dy+3)*7 + (dx+3), p in [0,HW)
__global__ void k_corr_lookup(const _Float16* __restrict__ c0, const _Float16* __restrict__ c1,
                              const _Float16* __restrict__ c2, const _Float16* __restrict__ c3,
                              const float* __restrict__ coords, float* __restrict__ feat,
                              int h, int w) {
  const int HW = h * w;
  int i = blockIdx.x * 256 + threadIdx.x;
  if (i >= 196 * HW) return;
  int j = i / HW, p = i - j * HW;
  int l = j / 49, t = j - l * 49;
  float dy = (float)(t / 7 - 3);
  float dx = (float)(t % 7 - 3);
  int hl = h >> l, wl = w >> l;
  const _Float16* lvl = (l == 0) ? c0 : (l == 1) ? c1 : (l == 2) ? c2 : c3;
  const _Float16* img = lvl + (size_t)p * hl * wl;
  float inv = 1.f / (float)(1 << l);
  float cx = coords[p] * inv + dx;
  float cy = coords[HW + p] * inv + dy;
  float x0 = floorf(cx), y0 = floorf(cy);
  float wx = cx - x0, wy = cy - y0;
  float v00 = samp_zero_h(img, x0,       y0,       hl, wl);
  float v01 = samp_zero_h(img, x0,       y0 + 1.f, hl, wl);
  float v10 = samp_zero_h(img, x0 + 1.f, y0,       hl, wl);
  float v11 = samp_zero_h(img, x0 + 1.f, y0 + 1.f, hl, wl);
  feat[i] = (1.f - wx) * (1.f - wy) * v00 + (1.f - wx) * wy * v01 +
            wx * (1.f - wy) * v10 + wx * wy * v11;
}

// 8x bilinear upsample (half-pixel centers, edge clamp), x8 scaling; two planes.
__global__ void k_upsample8(const float* __restrict__ flow, float* __restrict__ out,
                            int h, int w, int H8, int W8) {
  int i = blockIdx.x * 256 + threadIdx.x;
  int plane = H8 * W8;
  if (i >= 2 * plane) return;
  int c = i / plane;
  int r = i - c * plane;
  int Y = r / W8, X = r - Y * W8;
  float sy = ((float)Y + 0.5f) * 0.125f - 0.5f;
  float sx = ((float)X + 0.5f) * 0.125f - 0.5f;
  float y0 = floorf(sy), x0 = floorf(sx);
  float wy = sy - y0, wx = sx - x0;
  int yi0 = min(max((int)y0, 0), h - 1);
  int yi1 = min(max((int)y0 + 1, 0), h - 1);
  int xi0 = min(max((int)x0, 0), w - 1);
  int xi1 = min(max((int)x0 + 1, 0), w - 1);
  const float* img = flow + (size_t)c * h * w;
  float v = (1.f - wx) * (1.f - wy) * img[yi0 * w + xi0] +
            (1.f - wx) * wy         * img[yi1 * w + xi0] +
            wx * (1.f - wy)         * img[yi0 * w + xi1] +
            wx * wy                 * img[yi1 * w + xi1];
  out[i] = 8.f * v;
}

// ---------------------------------------------------------------------------
// Host orchestration
// ---------------------------------------------------------------------------
#define EW(kern, n, ...) kern<<<(((n) + 255) / 256), 256, 0, stream>>>(__VA_ARGS__)

extern "C" void kernel_launch(void* const* d_in, const int* in_sizes, int n_in,
                              void* d_out, int out_size, void* d_ws, size_t ws_size,
                              hipStream_t stream) {
  (void)in_sizes; (void)n_in; (void)out_size; (void)ws_size;
  const int H = 384, W = 1024;
  const int h = 48, w = 128, HW = h * w;

  const float* image1 = (const float*)d_in[0];
  const float* image2 = (const float*)d_in[1];
  const float* f_w1 = (const float*)d_in[2];  const float* f_b1 = (const float*)d_in[3];
  const float* f_w2 = (const float*)d_in[4];  const float* f_b2 = (const float*)d_in[5];
  const float* f_w3 = (const float*)d_in[6];  const float* f_b3 = (const float*)d_in[7];
  const float* c_w1 = (const float*)d_in[8];  const float* c_b1 = (const float*)d_in[9];
  const float* c_w2 = (const float*)d_in[10]; const float* c_b2 = (const float*)d_in[11];
  const float* c_w3 = (const float*)d_in[12]; const float* c_b3 = (const float*)d_in[13];
  const float* mc1_w = (const float*)d_in[14]; const float* mc1_b = (const float*)d_in[15];
  const float* mf1_w = (const float*)d_in[16]; const float* mf1_b = (const float*)d_in[17];
  const float* mf2_w = (const float*)d_in[18]; const float* mf2_b = (const float*)d_in[19];
  const float* m_w  = (const float*)d_in[20]; const float* m_b  = (const float*)d_in[21];
  const float* gz_w = (const float*)d_in[22]; const float* gz_b = (const float*)d_in[23];
  const float* gr_w = (const float*)d_in[24]; const float* gr_b = (const float*)d_in[25];
  const float* gq_w = (const float*)d_in[26]; const float* gq_b = (const float*)d_in[27];
  const float* fh1_w = (const float*)d_in[28]; const float* fh1_b = (const float*)d_in[29];
  const float* fh2_w = (const float*)d_in[30]; const float* fh2_b = (const float*)d_in[31];

  // ---- workspace carve-out (units: floats)
  float* WS = (float*)d_ws;
  size_t off = 0;
  auto alloc = [&](size_t n) { float* p = WS + off; off += n; return p; };
  float* img   = alloc((size_t)3 * H * W);
  float* e1    = alloc((size_t)64 * 192 * 512);
  float* e2    = alloc((size_t)96 * 96 * 256);
  float* fmap1 = alloc((size_t)128 * HW);
  float* fmap2 = alloc((size_t)128 * HW);
  float* cmap  = alloc((size_t)160 * HW);
  _Float16* f1h = (_Float16*)alloc((size_t)64 * HW);       // 128*HW halves
  _Float16* f2h = (_Float16*)alloc((size_t)64 * HW);
  _Float16* corr0 = (_Float16*)alloc((size_t)HW * HW / 2);        // f16 pyramid
  _Float16* corr1 = (_Float16*)alloc((size_t)HW * (HW / 4) / 2);
  _Float16* corr2 = (_Float16*)alloc((size_t)HW * (HW / 16) / 2);
  _Float16* corr3 = (_Float16*)alloc((size_t)HW * (HW / 64) / 2);
  float* coords = alloc((size_t)2 * HW);
  float* net    = alloc((size_t)96 * HW);
  float* x_in   = alloc((size_t)146 * HW);   // [mo:80 | flow:2 | ctx:64]
  float* feat   = alloc((size_t)196 * HW);
  float* cat128 = alloc((size_t)128 * HW);   // [cor:96 | flo:32]
  float* zbuf   = alloc((size_t)96 * HW);
  float* rbuf   = alloc((size_t)96 * HW);
  float* qbuf   = alloc((size_t)96 * HW);
  float* dh1    = alloc((size_t)128 * HW);
  float* delta  = alloc((size_t)2 * HW);
  float* flowp  = x_in + (size_t)80 * HW;
  float* ctxp   = x_in + (size_t)82 * HW;

  auto conv = [&](const float* in0, int C0, const float* in1,
                  const float* wgt, const float* b, float* o,
                  int Cin, int Cout, int K, int Hi, int Wi, int Ho, int Wo,
                  int stride, int act) {
    int pt = (Ho - 1) * stride + K - Hi; pt = pt > 0 ? pt / 2 : 0;
    int pl = (Wo - 1) * stride + K - Wi; pl = pl > 0 ? pl / 2 : 0;
    dim3 g((Ho * Wo) / 64, (Cout + 31) / 32);
    k_conv_wmma<<<g, 256, 0, stream>>>(in0, C0, in1, wgt, b, o, Cin, Cout,
                                       K, K, Hi, Wi, Ho, Wo, stride, pt, pl, act);
  };

  // ---- encoders -----------------------------------------------------------
  EW(k_preproc, 3 * H * W, image1, img, 3 * H * W);
  conv(img, 3, nullptr, f_w1, f_b1, e1, 3, 64, 7, 384, 1024, 192, 512, 2, 1);
  conv(e1, 64, nullptr, f_w2, f_b2, e2, 64, 96, 3, 192, 512, 96, 256, 2, 1);
  conv(e2, 96, nullptr, f_w3, f_b3, fmap1, 96, 128, 3, 96, 256, 48, 128, 2, 0);
  conv(img, 3, nullptr, c_w1, c_b1, e1, 3, 64, 7, 384, 1024, 192, 512, 2, 1);
  conv(e1, 64, nullptr, c_w2, c_b2, e2, 64, 96, 3, 192, 512, 96, 256, 2, 1);
  conv(e2, 96, nullptr, c_w3, c_b3, cmap, 96, 160, 3, 96, 256, 48, 128, 2, 0);
  EW(k_net_ctx, 160 * HW, cmap, net, ctxp, HW);
  EW(k_preproc, 3 * H * W, image2, img, 3 * H * W);
  conv(img, 3, nullptr, f_w1, f_b1, e1, 3, 64, 7, 384, 1024, 192, 512, 2, 1);
  conv(e1, 64, nullptr, f_w2, f_b2, e2, 64, 96, 3, 192, 512, 96, 256, 2, 1);
  conv(e2, 96, nullptr, f_w3, f_b3, fmap2, 96, 128, 3, 96, 256, 48, 128, 2, 0);

  // ---- correlation volume + pyramid (f16, L2-resident) --------------------
  EW(k_f32_to_f16, 128 * HW, fmap1, f1h, 128 * HW);
  EW(k_f32_to_f16, 128 * HW, fmap2, f2h, 128 * HW);
  {
    dim3 g(HW / 64, HW / 32);
    k_corr_wmma<<<g, 256, 0, stream>>>(f1h, f2h, corr0, HW, 0.08838834764831845f);
  }
  EW(k_avgpool2h, HW * (HW / 4),  corr0, corr1, HW, 48, 128);
  EW(k_avgpool2h, HW * (HW / 16), corr1, corr2, HW, 24, 64);
  EW(k_avgpool2h, HW * (HW / 64), corr2, corr3, HW, 12, 32);
  EW(k_initcoords, HW, coords, w, HW);

  // ---- GRU iterations -----------------------------------------------------
  for (int it = 0; it < ITERS_; ++it) {
    EW(k_corr_lookup, 196 * HW, corr0, corr1, corr2, corr3, coords, feat, h, w);
    EW(k_flow, HW, coords, flowp, w, HW);
    conv(feat, 196, nullptr, mc1_w, mc1_b, cat128, 196, 96, 1, h, w, h, w, 1, 1);
    conv(flowp, 2, nullptr, mf1_w, mf1_b, dh1, 2, 64, 7, h, w, h, w, 1, 1);
    conv(dh1, 64, nullptr, mf2_w, mf2_b, cat128 + (size_t)96 * HW, 64, 32, 3, h, w, h, w, 1, 1);
    conv(cat128, 128, nullptr, m_w, m_b, x_in, 128, 80, 3, h, w, h, w, 1, 1);
    conv(net, 96, x_in, gz_w, gz_b, zbuf, 242, 96, 3, h, w, h, w, 1, 2);
    conv(net, 96, x_in, gr_w, gr_b, rbuf, 242, 96, 3, h, w, h, w, 1, 2);
    EW(k_rmul, 96 * HW, rbuf, net, rbuf, 96 * HW);
    conv(rbuf, 96, x_in, gq_w, gq_b, qbuf, 242, 96, 3, h, w, h, w, 1, 3);
    EW(k_gru, 96 * HW, net, zbuf, qbuf, 96 * HW);
    conv(net, 96, nullptr, fh1_w, fh1_b, dh1, 96, 128, 3, h, w, h, w, 1, 1);
    conv(dh1, 128, nullptr, fh2_w, fh2_b, delta, 128, 2, 3, h, w, h, w, 1, 0);
    EW(k_addinto, 2 * HW, coords, delta, 2 * HW);
  }

  // ---- output: flow_low -> 8x bilinear upsample, planes (u, v) ------------
  EW(k_flow, HW, coords, delta, w, HW);
  EW(k_upsample8, 2 * H * W, delta, (float*)d_out, h, w, H, W);
}
